// SelfAttention_43422119362568
// MI455X (gfx1250) — matrix operand order repaired and checked by reading, not compiled, not run on hardware
//
#include <hip/hip_runtime.h>

typedef __bf16 bf16_t;
typedef __attribute__((ext_vector_type(8)))  bf16_t v8bf;
typedef __attribute__((ext_vector_type(16))) bf16_t v16bf;
typedef __attribute__((ext_vector_type(8)))  float  v8f;
typedef __attribute__((ext_vector_type(4)))  int    v4i;

#define LOG2E 1.44269504088896340736f

// ---------------------------------------------------------------------------
// CDNA5 async global->LDS copy (ASYNCcnt) with sync fallback.
// Probe-learned signature: (AS1 int4* global_src, AS3 int4* lds_dst, imm, imm)
// ---------------------------------------------------------------------------
#if defined(__has_builtin)
#if __has_builtin(__builtin_amdgcn_global_load_async_to_lds_b128)
#define USE_ASYNC_COPY 1
#endif
#endif

static __device__ __forceinline__ void cp16(void* l, const void* g) {
#if defined(USE_ASYNC_COPY)
    __builtin_amdgcn_global_load_async_to_lds_b128(
        (__attribute__((address_space(1))) v4i*)(void*)g,
        (__attribute__((address_space(3))) v4i*)(void*)l, 0, 0);
#else
    *(v8bf*)l = *(const v8bf*)g;
#endif
}

static __device__ __forceinline__ void wait_async() {
#if defined(USE_ASYNC_COPY)
#if __has_builtin(__builtin_amdgcn_s_wait_asynccnt)
    __builtin_amdgcn_s_wait_asynccnt(0);
#else
    asm volatile("s_wait_asynccnt 0x0" ::: "memory");
#endif
#endif
}

static __device__ __forceinline__ v16bf cat16(v8bf lo, v8bf hi) {
    return __builtin_shufflevector(lo, hi, 0,1,2,3,4,5,6,7,8,9,10,11,12,13,14,15);
}

// ---------------------------------------------------------------------------
// fp32 -> bf16 elementwise (grid-stride)
// ---------------------------------------------------------------------------
__global__ __launch_bounds__(256) void cvt_f32_bf16(const float* __restrict__ in,
                                                    bf16_t* __restrict__ out, long n) {
    long i = (long)blockIdx.x * blockDim.x + threadIdx.x;
    long stride = (long)gridDim.x * blockDim.x;
    for (; i < n; i += stride) out[i] = (bf16_t)in[i];
}

// fp32 [R][C] -> bf16 transposed [C][R]  (coalesced writes, output-major index)
__global__ __launch_bounds__(256) void trans_f32_bf16(const float* __restrict__ in,
                                                      bf16_t* __restrict__ out,
                                                      int R, int Cc) {
    long total = (long)R * Cc;
    long i = (long)blockIdx.x * blockDim.x + threadIdx.x;
    long stride = (long)gridDim.x * blockDim.x;
    for (; i < total; i += stride) {
        int c = (int)(i / R), r = (int)(i % R);
        out[i] = (bf16_t)in[(long)r * Cc + c];
    }
}

// V: fp32 [b][s][h][d] -> bf16 [b][h][d][s]  (coalesced writes along s)
__global__ __launch_bounds__(256) void vtrans_f32_bf16(const float* __restrict__ in,
                                                       bf16_t* __restrict__ out,
                                                       int S, int nKV, long total) {
    long i = (long)blockIdx.x * blockDim.x + threadIdx.x;
    long stride = (long)gridDim.x * blockDim.x;
    for (; i < total; i += stride) {
        int  s = (int)(i % S);
        long t = i / S;
        int  d = (int)(t % 64); t /= 64;
        int  h = (int)(t % nKV);
        int  b = (int)(t / nKV);
        out[i] = (bf16_t)in[(((long)b * S + s) * nKV + h) * 64 + d];
    }
}

// ---------------------------------------------------------------------------
// GEMM: C[M,N] = A[M,K] @ BT[N,K]^T + bias[N]   (bf16, fp32 accum/out)
// A row-major, BT = B pre-transposed (K-contiguous). Tile 128x128x32,
// 8 waves as 4x2, wave tile 32x64 -> 8 WMMA / K-step. Double-buffered async
// staging, unrolled x2 so buffer indices are compile-time. Requires K%64==0.
// ---------------------------------------------------------------------------
#define BM 128
#define BN 128
#define BK 32
__global__ __launch_bounds__(256) void gemm_bf16(const bf16_t* __restrict__ A,
                                                 const bf16_t* __restrict__ BT,
                                                 const float*  __restrict__ bias,
                                                 float* __restrict__ C,
                                                 int M, int N, int K) {
    __shared__ __attribute__((aligned(16))) bf16_t As[2][BM][BK];
    __shared__ __attribute__((aligned(16))) bf16_t Bs[2][BN][BK];

    const int tid  = threadIdx.x;
    const int lane = tid & 31, wave = tid >> 5;
    const int wm = wave & 3, wn = wave >> 2;           // 4 x 2
    const int l16 = lane & 15, lh = lane >> 4;
    const int bm = blockIdx.y * BM, bn = blockIdx.x * BN;

    // per-thread staging addresses (both tiles are 128 rows x 32 K-contiguous)
    const int srow = tid >> 1, shalf = (tid & 1) * 16;
    const bf16_t* aSrc = A  + (long)(bm + srow) * K + shalf;
    const bf16_t* bSrc = BT + (long)(bn + srow) * K + shalf;

    v8f acc[2][4] = {};

    auto stage = [&](int sbuf, int kk) {
        cp16(&As[sbuf][srow][shalf],     aSrc + kk);
        cp16(&As[sbuf][srow][shalf + 8], aSrc + kk + 8);
        cp16(&Bs[sbuf][srow][shalf],     bSrc + kk);
        cp16(&Bs[sbuf][srow][shalf + 8], bSrc + kk + 8);
    };

    auto compute = [&](int cbuf) {
        const int klo = lh ? 16 : 0;     // A frag: lanes<16 K 0..15, lanes>=16 K 16..31
        const int kb  = lh ? 16 : 0;     // B frag: lanes<16 K 0..15, lanes>=16 K 16..31
        v16bf afrag[2], bfrag[4];
#pragma unroll
        for (int i = 0; i < 2; ++i) {
            const bf16_t* p = &As[cbuf][wm * 32 + i * 16 + l16][0];
            afrag[i] = cat16(*(const v8bf*)(p + klo), *(const v8bf*)(p + 8 + klo));
        }
#pragma unroll
        for (int j = 0; j < 4; ++j) {
            const bf16_t* p = &Bs[cbuf][wn * 64 + j * 16 + l16][kb];
            bfrag[j] = cat16(*(const v8bf*)p, *(const v8bf*)(p + 8));
        }
#pragma unroll
        for (int i = 0; i < 2; ++i)
#pragma unroll
            for (int j = 0; j < 4; ++j)
                acc[i][j] = __builtin_amdgcn_wmma_f32_16x16x32_bf16(
                    false, afrag[i], false, bfrag[j], (short)0, acc[i][j], false, false);
    };

    stage(0, 0);
    int k0 = 0;
    for (; k0 < K - 2 * BK; k0 += 2 * BK) {
        wait_async(); __syncthreads();
        stage(1, k0 + BK);
        compute(0);
        wait_async(); __syncthreads();
        stage(0, k0 + 2 * BK);
        compute(1);
    }
    // tail: chunks k0 (buf0) and k0+BK (buf1)
    wait_async(); __syncthreads();
    stage(1, k0 + BK);
    compute(0);
    wait_async(); __syncthreads();
    compute(1);

    // epilogue: C layout VGPR r -> row r (lanes 0-15) / r+8 (lanes 16-31)
#pragma unroll
    for (int i = 0; i < 2; ++i)
#pragma unroll
        for (int j = 0; j < 4; ++j) {
            int colg = bn + wn * 64 + j * 16 + l16;
            float bv = bias[colg];
            float* cp = C + (long)(bm + wm * 32 + i * 16 + lh * 8) * N + colg;
#pragma unroll
            for (int r = 0; r < 8; ++r)
                cp[(long)r * N] = acc[i][j][r] + bv;
        }
}

// ---------------------------------------------------------------------------
// Fused RMSNorm (HEAD_DIM=64) + RoPE; one wave per (token, head).
// ---------------------------------------------------------------------------
__global__ __launch_bounds__(256) void rmsrope(const float* __restrict__ x,
                                               const float* __restrict__ w,
                                               const float* __restrict__ cosT,
                                               const float* __restrict__ sinT,
                                               bf16_t* __restrict__ out,
                                               int nHeads, int S, long nItems) {
    const int lane = threadIdx.x & 31;
    const int wv   = threadIdx.x >> 5;
    long item = (long)blockIdx.x * 8 + wv;
    if (item >= nItems) return;
    int token = (int)(item / nHeads), head = (int)(item % nHeads);

    const float* xp = x + (long)token * nHeads * 64 + head * 64;
    float lo = xp[lane], hi = xp[lane + 32];
    float ss = lo * lo + hi * hi;
#pragma unroll
    for (int m = 16; m >= 1; m >>= 1) ss += __shfl_xor(ss, m, 32);
    float inv = rsqrtf(ss * (1.0f / 64.0f) + 1e-6f);
    float nlo = lo * inv * w[lane];
    float nhi = hi * inv * w[lane + 32];

    int pos = token % S;
    const float* c = cosT + (long)pos * 64;
    const float* s = sinT + (long)pos * 64;
    float olo = nlo * c[lane]      - nhi * s[lane];
    float ohi = nhi * c[lane + 32] + nlo * s[lane + 32];

    bf16_t* op = out + (long)token * nHeads * 64 + head * 64;
    op[lane]      = (bf16_t)olo;
    op[lane + 32] = (bf16_t)ohi;
}

// ---------------------------------------------------------------------------
// Flash attention (causal, GQA). Block = (qtile, head, batch), 8 waves,
// wave owns 16 q rows. KV in blocks of 32 via double-buffered async LDS
// staging, unrolled x2 (kvEnd % 64 == 0). V pre-transposed to [b][h][d][S].
// ---------------------------------------------------------------------------
#define KBLK 32
__global__ __launch_bounds__(256) void flash_attn(const bf16_t* __restrict__ Q,
                                                  const bf16_t* __restrict__ Kc,
                                                  const bf16_t* __restrict__ VtG,
                                                  bf16_t* __restrict__ O,
                                                  int S, int nHeads, int nKV) {
    __shared__ __attribute__((aligned(16))) bf16_t Ks[2][KBLK][64];
    __shared__ __attribute__((aligned(16))) bf16_t Vs[2][64][KBLK];   // [d][k]
    __shared__ __attribute__((aligned(16))) bf16_t Ps[8][16][KBLK];

    const int tid  = threadIdx.x;
    const int lane = tid & 31, wave = tid >> 5;
    const int l16 = lane & 15, lh = lane >> 4;
    const int qt = blockIdx.x, head = blockIdx.y, batch = blockIdx.z;
    const int kvh = head / (nHeads / nKV);
    const int sq0 = qt * 128 + wave * 16;
    const long qRow0 = (long)batch * S + sq0;

    // staging addresses
    const int kr = tid >> 3, kc = (tid & 7) * 8;          // K: 32 rows x 64
    const bf16_t* kBase = Kc + ((long)batch * S + kr) * (long)nKV * 64 + (long)kvh * 64 + kc;
    const int vd = tid >> 2, vc = (tid & 3) * 8;          // V^T: 64 rows x 32
    const bf16_t* vBase = VtG + (((long)(batch * nKV + kvh)) * 64 + vd) * S + vc;

    auto stage = [&](int sbuf, int kb) {
        cp16(&Ks[sbuf][kr][kc], kBase + (long)kb * nKV * 64);
        cp16(&Vs[sbuf][vd][vc], vBase + kb);
    };

    // Q A-fragments (16x64, two K-steps of 32 along head_dim)
    v16bf qfrag[2];
    {
        const bf16_t* qp = Q + (qRow0 + l16) * (long)nHeads * 64 + (long)head * 64;
        const int klo = lh ? 16 : 0;
#pragma unroll
        for (int i = 0; i < 2; ++i) {
            const bf16_t* p = qp + i * 32;
            qfrag[i] = cat16(*(const v8bf*)(p + klo), *(const v8bf*)(p + 8 + klo));
        }
    }

    v8f acc[4] = {};
    float m_r[8], l_r[8];
#pragma unroll
    for (int r = 0; r < 8; ++r) { m_r[r] = -1e30f; l_r[r] = 0.f; }

    const float sscale = 0.125f * LOG2E;
    const int kvEnd = qt * 128 + 128;

    auto compute = [&](int cbuf, int kb) {
        // scores: Q[16x64] . K^T[64x16] for two 16-col tiles
        v8f sv[2];
#pragma unroll
        for (int ct = 0; ct < 2; ++ct) {
            v8f s = {};
#pragma unroll
            for (int dstep = 0; dstep < 2; ++dstep) {
                const bf16_t* p = &Ks[cbuf][ct * 16 + l16][dstep * 32 + (lh ? 16 : 0)];
                v16bf bf = cat16(*(const v8bf*)p, *(const v8bf*)(p + 8));
                s = __builtin_amdgcn_wmma_f32_16x16x32_bf16(
                    false, qfrag[dstep], false, bf, (short)0, s, false, false);
            }
            sv[ct] = s;
        }
        // scale + causal mask (log2 domain)
#pragma unroll
        for (int ct = 0; ct < 2; ++ct) {
            int sk = kb + ct * 16 + l16;
#pragma unroll
            for (int r = 0; r < 8; ++r) {
                int sq = sq0 + lh * 8 + r;
                float vsc = sv[ct][r] * sscale;
                sv[ct][r] = (sk <= sq) ? vsc : -1e30f;
            }
        }
        // online softmax (row reductions within 16-lane groups)
        float rx[8];
#pragma unroll
        for (int r = 0; r < 8; ++r) rx[r] = fmaxf(sv[0][r], sv[1][r]);
#pragma unroll
        for (int m = 8; m >= 1; m >>= 1)
#pragma unroll
            for (int r = 0; r < 8; ++r) rx[r] = fmaxf(rx[r], __shfl_xor(rx[r], m, 32));

        float alpha[8];
#pragma unroll
        for (int r = 0; r < 8; ++r) {
            float mn = fmaxf(m_r[r], rx[r]);
            alpha[r] = exp2f(m_r[r] - mn);
            m_r[r] = mn;
        }
        float rs[8];
#pragma unroll
        for (int r = 0; r < 8; ++r) rs[r] = 0.f;
#pragma unroll
        for (int ct = 0; ct < 2; ++ct)
#pragma unroll
            for (int r = 0; r < 8; ++r) {
                float p = exp2f(sv[ct][r] - m_r[r]);
                sv[ct][r] = p;
                rs[r] += p;
            }
#pragma unroll
        for (int m = 8; m >= 1; m >>= 1)
#pragma unroll
            for (int r = 0; r < 8; ++r) rs[r] += __shfl_xor(rs[r], m, 32);
#pragma unroll
        for (int r = 0; r < 8; ++r) l_r[r] = l_r[r] * alpha[r] + rs[r];
#pragma unroll
        for (int t = 0; t < 4; ++t)
#pragma unroll
            for (int r = 0; r < 8; ++r) acc[t][r] *= alpha[r];

        // P (C-layout) -> LDS -> A-fragment; P[16x32] . V[32x64]
#pragma unroll
        for (int ct = 0; ct < 2; ++ct)
#pragma unroll
            for (int r = 0; r < 8; ++r)
                Ps[wave][lh * 8 + r][ct * 16 + l16] = (bf16_t)sv[ct][r];
        v16bf pfrag;
        {
            const int klo = lh ? 16 : 0;
            const bf16_t* p = &Ps[wave][l16][0];
            pfrag = cat16(*(const v8bf*)(p + klo), *(const v8bf*)(p + 8 + klo));
        }
#pragma unroll
        for (int nt = 0; nt < 4; ++nt) {
            const bf16_t* p = &Vs[cbuf][nt * 16 + l16][lh ? 16 : 0];
            v16bf vfrag = cat16(*(const v8bf*)p, *(const v8bf*)(p + 8));
            acc[nt] = __builtin_amdgcn_wmma_f32_16x16x32_bf16(
                false, pfrag, false, vfrag, (short)0, acc[nt], false, false);
        }
    };

    stage(0, 0);
    for (int kb = 0; kb < kvEnd; kb += 2 * KBLK) {
        wait_async(); __syncthreads();
        stage(1, kb + KBLK);                              // kb+KBLK < kvEnd always
        if (kb <= sq0 + 15) compute(0, kb);
        wait_async(); __syncthreads();
        if (kb + 2 * KBLK < kvEnd) stage(0, kb + 2 * KBLK);
        if (kb + KBLK <= sq0 + 15) compute(1, kb + KBLK);
    }

    // epilogue: normalize, store bf16 for O-projection
#pragma unroll
    for (int nt = 0; nt < 4; ++nt)
#pragma unroll
        for (int r = 0; r < 8; ++r) {
            long row = qRow0 + lh * 8 + r;
            O[row * (long)nHeads * 64 + (long)head * 64 + nt * 16 + l16] =
                (bf16_t)(acc[nt][r] / l_r[r]);
        }
}

// ---------------------------------------------------------------------------
extern "C" void kernel_launch(void* const* d_in, const int* in_sizes, int n_in,
                              void* d_out, int out_size, void* d_ws, size_t ws_size,
                              hipStream_t stream) {
    const float* hs   = (const float*)d_in[0];
    const float* Wq   = (const float*)d_in[1];
    const float* bq   = (const float*)d_in[2];
    const float* Wk   = (const float*)d_in[3];
    const float* bk   = (const float*)d_in[4];
    const float* Wv   = (const float*)d_in[5];
    const float* bv   = (const float*)d_in[6];
    const float* Wo   = (const float*)d_in[7];
    const float* bo   = (const float*)d_in[8];
    const float* qnw  = (const float*)d_in[9];
    const float* knw  = (const float*)d_in[10];
    const float* cosT = (const float*)d_in[11];
    const float* sinT = (const float*)d_in[12];

    const int HID = 2048, NH = 32, NKV = 8, HD = 64;
    const int S  = in_sizes[11] / HD;
    const int T  = in_sizes[0] / HID;
    const int Bb = T / S;
    const int NQ = NH * HD;            // 2048
    const int NK = NKV * HD;           // 512

    char* w = (char*)d_ws;
    auto carve = [&](size_t bytes) { char* p = w; w += (bytes + 255) & ~(size_t)255; return p; };
    bf16_t* hsb = (bf16_t*)carve((size_t)T * HID * 2);
    bf16_t* wqt = (bf16_t*)carve((size_t)HID * NQ * 2);   // [NQ][HID]
    bf16_t* wkt = (bf16_t*)carve((size_t)HID * NK * 2);   // [NK][HID]
    bf16_t* wvt = (bf16_t*)carve((size_t)HID * NK * 2);
    bf16_t* wot = (bf16_t*)carve((size_t)NQ * HID * 2);   // [HID][NQ]
    float*  qf  = (float*)carve((size_t)T * NQ * 4);
    float*  kf  = (float*)carve((size_t)T * NK * 4);
    float*  vf  = (float*)carve((size_t)T * NK * 4);
    // aliases (producers dead by write time)
    bf16_t* qb  = (bf16_t*)hsb;                              // after projections
    bf16_t* kb  = (bf16_t*)wqt;
    bf16_t* vtb = (bf16_t*)((char*)wqt + (size_t)T * NK * 2);
    bf16_t* ob  = (bf16_t*)qf;                               // after rmsrope(q)

    auto cvt = [&](const float* in, bf16_t* out, long n) {
        int g = (int)((n + 2047) / 2048); if (g > 8192) g = 8192;
        cvt_f32_bf16<<<g, 256, 0, stream>>>(in, out, n);
    };
    auto trans = [&](const float* in, bf16_t* out, int R, int Cc) {
        long n = (long)R * Cc;
        int g = (int)((n + 2047) / 2048); if (g > 8192) g = 8192;
        trans_f32_bf16<<<g, 256, 0, stream>>>(in, out, R, Cc);
    };

    // 1) staging: activations bf16; weights bf16 *transposed* (K-contiguous)
    cvt(hs, hsb, (long)T * HID);
    trans(Wq, wqt, HID, NQ);
    trans(Wk, wkt, HID, NK);
    trans(Wv, wvt, HID, NK);
    trans(Wo, wot, NQ, HID);

    // 2) QKV projections (WMMA bf16, fp32 accum)
    gemm_bf16<<<dim3(NQ / BN, T / BM), 256, 0, stream>>>(hsb, wqt, bq, qf, T, NQ, HID);
    gemm_bf16<<<dim3(NK / BN, T / BM), 256, 0, stream>>>(hsb, wkt, bk, kf, T, NK, HID);
    gemm_bf16<<<dim3(NK / BN, T / BM), 256, 0, stream>>>(hsb, wvt, bv, vf, T, NK, HID);

    // 3) RMSNorm + RoPE -> bf16 Q/K ; V -> bf16 transposed [b][h][d][s]
    rmsrope<<<(int)(((long)T * NH) / 8), 256, 0, stream>>>(qf, qnw, cosT, sinT, qb, NH, S, (long)T * NH);
    rmsrope<<<(int)(((long)T * NKV) / 8), 256, 0, stream>>>(kf, knw, cosT, sinT, kb, NKV, S, (long)T * NKV);
    {
        long n = (long)T * NK;
        int g = (int)((n + 2047) / 2048); if (g > 8192) g = 8192;
        vtrans_f32_bf16<<<g, 256, 0, stream>>>(vf, vtb, S, NKV, n);
    }

    // 4) causal flash attention (GQA 4:1)
    flash_attn<<<dim3(S / 128, NH, Bb), 256, 0, stream>>>(qb, kb, vtb, ob, S, NH, NKV);

    // 5) output projection -> fp32 d_out
    gemm_bf16<<<dim3(HID / BN, T / BM), 256, 0, stream>>>(ob, wot, bo, (float*)d_out, T, HID, NQ);
}
